// DepthConv_67044439491174
// MI455X (gfx1250) — compile-verified, matrix operand order checked by reading, not run on hardware
//
#include <hip/hip_runtime.h>

// CDNA5 / gfx1250 WMMA implicit-GEMM depth-modulated 3x3 conv.
// Per 16x16 tile: D(f32) += A(weight 16x32 f16) x B(dw-modulated img 32x16 f16)
// Block = 256 threads (8 wave32) = 64 output cols x 4 output rows x 64 OC.

typedef __attribute__((ext_vector_type(16))) _Float16 v16h;
typedef __attribute__((ext_vector_type(8)))  float    v8f;

#define DC_ALPHA 8.3f

__device__ __forceinline__ unsigned int pack2h(float a, float b) {
  // single v_cvt_pk_rtz_f16_f32
  return __builtin_bit_cast(unsigned int, __builtin_amdgcn_cvt_pkrtz(a, b));
}

// Load one 16x32 f16 WMMA operand fragment from LDS.
// Per row/col: K pairs contiguous (u32), 16B-aligned base, padded row stride.
// ISA layout (CDNA5 ISA 7.12.2): lane half 0 -> K2 {kt*16+0..3, kt*16+8..11},
// lane half 1 -> K2 {kt*16+4..7, kt*16+12..15}; two ds_load_b128 per fragment.
__device__ __forceinline__ v16h load_frag(const unsigned int* base, int kt, int lh) {
  union { uint4 q[2]; v16h h; } f;
  const unsigned int* p = base + kt * 16 + lh * 4;
  f.q[0] = *reinterpret_cast<const uint4*>(p);
  f.q[1] = *reinterpret_cast<const uint4*>(p + 8);
  return f.h;
}

#define ASTR 292  // u32 stride per OC row in aSh: 9 taps * 32 pairs + pad (16B-aligned, bank walk)
#define BSTR 36   // u32 stride per pixel col in bSh (16B-aligned, bank walk)

// Stage one tap of B: 64 px x 32 channel-pairs of dw-modulated img as f16 pairs.
// Each iteration handles a channel QUAD (2 pairs) -> one ds_store_b64.
__device__ __forceinline__ void stage_B(unsigned int* __restrict__ dst,
                                        const float* __restrict__ img_b,
                                        const float* __restrict__ dwrow,
                                        int p, int q0, int kk, int tid) {
  const int k = kk / 3, l = kk - 3 * k;
  const float* iplane = img_b + (size_t)(p + k) * 256;
  #pragma unroll
  for (int i = 0; i < 4; ++i) {
    int idx = i * 256 + tid;           // 1024 quad-items
    int th  = idx >> 6;                // 0..15 -> channels c = 4*th .. 4*th+3
    int n   = idx & 63;                // consecutive lanes -> consecutive n (coalesced)
    int qi = q0 + n + l; if (qi > 255) qi = 255;
    float dwv = dwrow[n];
    const float* ip = iplane + (size_t)(4 * th) * 65536 + qi;
    float v0 = ip[0 * 65536];
    float v1 = ip[1 * 65536];
    float v2 = ip[2 * 65536];
    float v3 = ip[3 * 65536];
    uint2 pr;
    pr.x = pack2h(v0 * dwv, v1 * dwv);
    pr.y = pack2h(v2 * dwv, v3 * dwv);
    *reinterpret_cast<uint2*>(&dst[n * BSTR + 2 * th]) = pr;  // ds_store_b64
  }
}

__global__ __launch_bounds__(256) void depthconv_wmma_kernel(
    const float* __restrict__ img,     // (4,64,256,256)
    const float* __restrict__ depth,   // (4,1,256,256)
    const float* __restrict__ weight,  // (64,64,3,3)
    const float* __restrict__ bias,    // (1,64)
    float* __restrict__ out) {         // (4,64,254,254)
  constexpr int W = 256, OW = 254, OH = 254;

  __shared__ __align__(16) unsigned int aSh[64 * ASTR];     // all-tap weight fragments (~73 KB)
  __shared__ __align__(16) unsigned int bSh[2][64 * BSTR];  // double-buffered img fragments
  __shared__ float dwSh[9 * 64];                            // depth weights (tap x px)

  const int tid = threadIdx.x;
  const int q0  = blockIdx.x * 64;       // first output column
  const int p0  = blockIdx.y * 4;        // first output row
  const int b   = blockIdx.z;            // batch

  const int wv   = tid >> 5;             // wave id (wave32)
  const int lane = tid & 31;
  const int lrow = lane & 15;
  const int lh   = lane >> 4;
  const int mt   = wv & 3;               // OC tile
  const int ntb  = (wv >> 2) * 2;        // first of two pixel tiles

  const float* img_b  = img + (size_t)b * 64 * (W * W);
  const float* dplane = depth + (size_t)b * (W * W);

  // ---- stage A once for all 9 taps: pairs (c even, c odd) at K2 = kk*32 + t ----
  #pragma unroll 4
  for (int idx = tid; idx < 64 * 288; idx += 256) {
    int o = idx / 288;
    int rest = idx - o * 288;            // = kk*32 + t
    int kk = rest >> 5, t = rest & 31;
    float w0 = weight[o * 576 + (2 * t) * 9 + kk];
    float w1 = weight[o * 576 + (2 * t + 1) * 9 + kk];
    aSh[o * ASTR + rest] = pack2h(w0, w1);
  }

  float bl[8];
  #pragma unroll
  for (int v = 0; v < 8; ++v) bl[v] = bias[mt * 16 + 8 * lh + v];

  for (int r = 0; r < 4; ++r) {
    const int p = p0 + r;
    if (p >= OH) break;                  // uniform across block

    // ---- stage dw[kk][n] = exp(-alpha*|depth[p+k][q+l] - depth[p+1][q+1]|) ----
    for (int idx = tid; idx < 576; idx += 256) {
      int kk = idx >> 6, n = idx & 63;
      int k = kk / 3, l = kk - 3 * k;
      int q  = q0 + n;
      int qc = q + 1; if (qc > 255) qc = 255;  // clamp only for unused pad lanes
      int qi = q + l; if (qi > 255) qi = 255;
      float c = dplane[(p + 1) * W + qc];
      float d = dplane[(p + k) * W + qi];
      dwSh[idx] = __expf(-DC_ALPHA * fabsf(d - c));
    }
    __syncthreads();                     // A (first row) + dw visible

    stage_B(bSh[0], img_b, dwSh, p, q0, 0, tid);
    __syncthreads();

    v8f acc0 = {};
    v8f acc1 = {};

    for (int kk = 0; kk < 9; ++kk) {
      if (kk < 8)                        // stage next tap while WMMA eats this one
        stage_B(bSh[(kk + 1) & 1], img_b, dwSh + (kk + 1) * 64, p, q0, kk + 1, tid);

      const unsigned int* arow = aSh + (mt * 16 + lrow) * ASTR + kk * 32;
      const unsigned int* bbuf = bSh[kk & 1];
      const unsigned int* bc0  = bbuf + (ntb * 16 + lrow) * BSTR;
      const unsigned int* bc1  = bbuf + ((ntb + 1) * 16 + lrow) * BSTR;
      #pragma unroll
      for (int kt = 0; kt < 2; ++kt) {   // K = 64 channels = 2 x 32
        v16h af = load_frag(arow, kt, lh);
        v16h b0 = load_frag(bc0, kt, lh);
        v16h b1 = load_frag(bc1, kt, lh);
        acc0 = __builtin_amdgcn_wmma_f32_16x16x32_f16(
            false, af, false, b0, (short)0, acc0, false, false);
        acc1 = __builtin_amdgcn_wmma_f32_16x16x32_f16(
            false, af, false, b1, (short)0, acc1, false, false);
      }
      __syncthreads();
    }

    // ---- epilogue: C/D layout M = vgpr + 8*lanehalf, N = lane%16 ----
    #pragma unroll
    for (int tIdx = 0; tIdx < 2; ++tIdx) {
      int nt = ntb + tIdx;
      int q  = q0 + nt * 16 + lrow;
      if (q < OW) {
        v8f acc = tIdx ? acc1 : acc0;
        #pragma unroll
        for (int v = 0; v < 8; ++v) {
          int o = mt * 16 + 8 * lh + v;
          out[(((size_t)b * 64 + o) * OH + p) * OW + q] = acc[v] + bl[v];
        }
      }
    }
  }
}

extern "C" void kernel_launch(void* const* d_in, const int* in_sizes, int n_in,
                              void* d_out, int out_size, void* d_ws, size_t ws_size,
                              hipStream_t stream) {
  (void)in_sizes; (void)n_in; (void)out_size; (void)d_ws; (void)ws_size;
  const float* img    = (const float*)d_in[0];
  const float* depth  = (const float*)d_in[1];
  const float* weight = (const float*)d_in[2];
  const float* bias   = (const float*)d_in[3];
  float* out = (float*)d_out;

  dim3 grid(4 /* 64-col blocks */, 64 /* 4-row blocks over oh=254 */, 4 /* batch */);
  dim3 block(256);
  depthconv_wmma_kernel<<<grid, block, 0, stream>>>(img, depth, weight, bias, out);
}
